// HoloLinear_19301583029003
// MI455X (gfx1250) — matrix operand order repaired and checked by reading, not compiled, not run on hardware
//
#include <hip/hip_runtime.h>

// HoloLinear: out[16384,8192] = (X[16384,2048] @ basis.T) @ (amp*cos(phase)).T
// bf16 WMMA (v_wmma_f32_16x16x32_bf16), f32 accumulate. Pipeline:
//   k0a: x_bf16     = bf16(x)               (if ws allows; else convert in gemm1)
//   k0b: basis_bf16 = bf16(basis)
//   k1 : w_bf16     = bf16(amp*cos(phase))
//   k2 : r_bf16     = x @ basis.T    (basis slice staged in LDS, double-buffered)
//   k3 : out        = r_bf16 @ w_bf16.T  (block's A slice staged in LDS once)
// Roofline: 86 GFLOP, ~0.74 GB HBM -> ~32us @ 23.3 TB/s. LDS staging cuts L2
// reads from ~2.1 GB to ~1.2 GB so the 536 MB output stream stays the limiter.

typedef __attribute__((ext_vector_type(16))) __bf16 v16bf;
typedef __attribute__((ext_vector_type(8)))  float  v8f;

#define IN_F  2048
#define HARM  256
#define OUT_F 8192
#define ROWS  16384  // 4*4096

static __device__ __forceinline__ unsigned short f32_to_bf16(float f) {
    unsigned int u = __float_as_uint(f);
    u += 0x7FFFu + ((u >> 16) & 1u);     // round-to-nearest-even
    return (unsigned short)(u >> 16);
}

union Frag16 {
    uint4          q[2];
    unsigned short s[16];
    v16bf          v;
};

// ---------------- elementwise converters ----------------
__global__ void __launch_bounds__(256)
cvt_f32_bf16_kernel(const float* __restrict__ src, unsigned short* __restrict__ dst) {
    const int i = (blockIdx.x * 256 + threadIdx.x) * 4;
    const float4 f = *(const float4*)(src + i);
    ushort4 r;
    r.x = f32_to_bf16(f.x); r.y = f32_to_bf16(f.y);
    r.z = f32_to_bf16(f.z); r.w = f32_to_bf16(f.w);
    *(ushort4*)(dst + i) = r;
}

__global__ void __launch_bounds__(256)
wreal_kernel(const float* __restrict__ phase, const float* __restrict__ amp,
             unsigned short* __restrict__ wbf) {
    const int i = (blockIdx.x * 256 + threadIdx.x) * 4;
    const float4 p = *(const float4*)(phase + i);
    const float4 a = *(const float4*)(amp + i);
    ushort4 r;
    r.x = f32_to_bf16(a.x * cosf(p.x));
    r.y = f32_to_bf16(a.y * cosf(p.y));
    r.z = f32_to_bf16(a.z * cosf(p.z));
    r.w = f32_to_bf16(a.w * cosf(p.w));
    *(ushort4*)(wbf + i) = r;
}

// ---------------- k2: r_bf16[ROWS,HARM] = X @ basis.T ----------------
// Block: 8 waves x 32 rows = 256 rows, shared 128-harmonic N range.
// Per k-step the 128x32 basis slice (8 KB) is staged in double-buffered LDS
// (row stride 40 elems = 80 B, 16B-aligned, spreads banks), so the block loads
// basis once instead of once per wave.
// A layout (16x32 bf16): lane = row (l15); lanes 0-15 hold K {0-7,16-23},
// lanes 16-31 hold K {8-15,24-31}. B layout (32x16): lane = col; same K split as A.
#define B_STRIDE 40
#define B_BUF    (128 * B_STRIDE)

template <bool XBF>
__global__ void __launch_bounds__(256)
gemm1_kernel(const float* __restrict__ xf, const unsigned short* __restrict__ xbf,
             const unsigned short* __restrict__ basisbf,
             unsigned short* __restrict__ rbf) {
    __shared__ unsigned short Bs[2 * B_BUF];

    const int lane = threadIdx.x & 31;
    const int wave = threadIdx.x >> 5;
    const int half = lane >> 4;
    const int l15  = lane & 15;

    const int mbase = (blockIdx.x * 8 + wave) * 32;
    const int nbase = blockIdx.y * 128;

    v8f acc[2][8];
    #pragma unroll
    for (int a = 0; a < 2; ++a)
        #pragma unroll
        for (int t = 0; t < 8; ++t)
            acc[a][t] = (v8f){0.f,0.f,0.f,0.f,0.f,0.f,0.f,0.f};

    // stage k-slice `ks` of basis into buffer `buf`: 512 uint4, 2 per thread
    auto stage = [&](int ks, int buf) {
        const int k0 = ks * 32;
        #pragma unroll
        for (int i = 0; i < 2; ++i) {
            const int u  = threadIdx.x + i * 256;   // 0..511
            const int n  = u >> 2;                  // 0..127
            const int c8 = (u & 3) * 8;             // 0,8,16,24
            *(uint4*)(&Bs[buf * B_BUF + n * B_STRIDE + c8]) =
                *(const uint4*)(basisbf + (size_t)(nbase + n) * IN_F + k0 + c8);
        }
    };

    stage(0, 0);
    int cur = 0;

    for (int ks = 0; ks < IN_F / 32; ++ks) {
        __syncthreads();
        if (ks + 1 < IN_F / 32) stage(ks + 1, cur ^ 1);

        const int k0 = ks * 32;
        Frag16 A[2];
        #pragma unroll
        for (int a = 0; a < 2; ++a) {
            const int mrow = mbase + a * 16 + l15;
            if (XBF) {
                const unsigned short* ar = xbf + (size_t)mrow * IN_F + k0 + half * 8;
                A[a].q[0] = ((const uint4*)ar)[0];
                A[a].q[1] = *(const uint4*)(ar + 16);
            } else {
                const float* ar = xf + (size_t)mrow * IN_F + k0 + half * 8;
                const float4 f0 = ((const float4*)ar)[0];
                const float4 f1 = ((const float4*)ar)[1];
                const float4 f2 = ((const float4*)(ar + 16))[0];
                const float4 f3 = ((const float4*)(ar + 16))[1];
                const float fv[16] = {f0.x,f0.y,f0.z,f0.w, f1.x,f1.y,f1.z,f1.w,
                                      f2.x,f2.y,f2.z,f2.w, f3.x,f3.y,f3.z,f3.w};
                #pragma unroll
                for (int j = 0; j < 16; ++j) A[a].s[j] = f32_to_bf16(fv[j]);
            }
        }
        #pragma unroll
        for (int t = 0; t < 8; ++t) {
            const unsigned short* bp =
                &Bs[cur * B_BUF + (t * 16 + l15) * B_STRIDE + half * 8];
            Frag16 B;
            B.q[0] = ((const uint4*)bp)[0];
            B.q[1] = *(const uint4*)(bp + 16);
            #pragma unroll
            for (int a = 0; a < 2; ++a)
                acc[a][t] = __builtin_amdgcn_wmma_f32_16x16x32_bf16(
                    false, A[a].v, false, B.v, (short)0, acc[a][t], false, false);
        }
        cur ^= 1;
    }

    // D layout: lane = N col (l15); VGPR r -> M = half*8 + r
    #pragma unroll
    for (int a = 0; a < 2; ++a)
        #pragma unroll
        for (int t = 0; t < 8; ++t) {
            const int ncol = nbase + t * 16 + l15;
            #pragma unroll
            for (int r = 0; r < 8; ++r) {
                const int m = mbase + a * 16 + half * 8 + r;
                rbf[(size_t)m * HARM + ncol] = f32_to_bf16(acc[a][t][r]);
            }
        }
}

// ---------------- k3: out[ROWS,OUT_F] = r_bf @ w_bf.T ----------------
// Block: 8 waves -> 64 rows x 512 cols. Only rows 0-31 of the block's A slice (32x256 bf16 =
// 16 KB) is staged once into padded LDS (row stride 264 elems = 528 B), then all
// waves read fragments with ds_load_b128 -> L2 reads 272 KB/block vs 128 KB out.
#define A_STRIDE 264

__global__ void __launch_bounds__(256)
gemm2_kernel(const unsigned short* __restrict__ rbf,
             const unsigned short* __restrict__ wbf,
             float* __restrict__ out) {
    __shared__ unsigned short As[64 * A_STRIDE];

    const int lane = threadIdx.x & 31;
    const int wave = threadIdx.x >> 5;
    const int half = lane >> 4;
    const int l15  = lane & 15;

    const int mbase = blockIdx.y * 64;
    const int nbase = blockIdx.x * 512 + wave * 64;

    // cooperative stage: 1024 uint4, 4 per thread
    #pragma unroll
    for (int i = 0; i < 4; ++i) {
        const int u   = threadIdx.x + i * 256;  // 0..1023
        const int row = u >> 5;                 // 0..31
        const int c8  = (u & 31) * 8;           // 0..248
        *(uint4*)(&As[row * A_STRIDE + c8]) =
            *(const uint4*)(rbf + (size_t)(mbase + row) * HARM + c8);
    }
    __syncthreads();

    v8f acc[4][4];
    #pragma unroll
    for (int a = 0; a < 4; ++a)
        #pragma unroll
        for (int t = 0; t < 4; ++t)
            acc[a][t] = (v8f){0.f,0.f,0.f,0.f,0.f,0.f,0.f,0.f};

    for (int ks = 0; ks < HARM / 32; ++ks) {
        const int k0 = ks * 32;
        Frag16 A[4];
        #pragma unroll
        for (int a = 0; a < 4; ++a) {
            const unsigned short* ap =
                &As[(a * 16 + l15) * A_STRIDE + k0 + half * 8];
            A[a].q[0] = ((const uint4*)ap)[0];
            A[a].q[1] = *(const uint4*)(ap + 16);
        }
        #pragma unroll
        for (int t = 0; t < 4; ++t) {
            const unsigned short* br =
                wbf + (size_t)(nbase + t * 16 + l15) * HARM + k0 + half * 8;
            Frag16 B;
            B.q[0] = ((const uint4*)br)[0];
            B.q[1] = *(const uint4*)(br + 16);
            #pragma unroll
            for (int a = 0; a < 4; ++a)
                acc[a][t] = __builtin_amdgcn_wmma_f32_16x16x32_bf16(
                    false, A[a].v, false, B.v, (short)0, acc[a][t], false, false);
        }
    }

    #pragma unroll
    for (int a = 0; a < 4; ++a)
        #pragma unroll
        for (int t = 0; t < 4; ++t) {
            const int ncol = nbase + t * 16 + l15;
            #pragma unroll
            for (int r = 0; r < 8; ++r) {
                const int m = mbase + a * 16 + half * 8 + r;
                out[(size_t)m * OUT_F + ncol] = acc[a][t][r];
            }
        }
}

extern "C" void kernel_launch(void* const* d_in, const int* in_sizes, int n_in,
                              void* d_out, int out_size, void* d_ws, size_t ws_size,
                              hipStream_t stream) {
    const float* x     = (const float*)d_in[0];
    const float* basis = (const float*)d_in[1];
    const float* phase = (const float*)d_in[2];
    const float* amp   = (const float*)d_in[3];
    float* out = (float*)d_out;

    // workspace layout (bytes):
    //   [0,   4Mi)  w_bf16     OUT_F*HARM
    //   [4Mi, 12Mi) r_bf16     ROWS*HARM
    //   [12Mi,13Mi) basis_bf16 HARM*IN_F
    //   [13Mi,77Mi) x_bf16     ROWS*IN_F    (optional tier)
    char* ws = (char*)d_ws;
    unsigned short* wbf = (unsigned short*)ws;
    unsigned short* rbf = (unsigned short*)(ws + (size_t)OUT_F * HARM * 2);
    unsigned short* bbf = (unsigned short*)(ws + (size_t)(OUT_F + ROWS) * HARM * 2);
    unsigned short* xbf = (unsigned short*)(ws + (size_t)(OUT_F + ROWS) * HARM * 2
                                               + (size_t)HARM * IN_F * 2);
    const size_t need_full = (size_t)(OUT_F + ROWS) * HARM * 2
                           + (size_t)HARM * IN_F * 2
                           + (size_t)ROWS * IN_F * 2;
    const bool use_xbf = (ws_size >= need_full);

    wreal_kernel<<<dim3((OUT_F * HARM) / 1024), dim3(256), 0, stream>>>(phase, amp, wbf);
    cvt_f32_bf16_kernel<<<dim3((HARM * IN_F) / 1024), dim3(256), 0, stream>>>(basis, bbf);
    if (use_xbf) {
        cvt_f32_bf16_kernel<<<dim3((ROWS * IN_F) / 1024), dim3(256), 0, stream>>>(x, xbf);
        gemm1_kernel<true><<<dim3(ROWS / 256, HARM / 128), dim3(256), 0, stream>>>(
            x, xbf, bbf, rbf);
    } else {
        gemm1_kernel<false><<<dim3(ROWS / 256, HARM / 128), dim3(256), 0, stream>>>(
            x, xbf, bbf, rbf);
    }
    gemm2_kernel<<<dim3(OUT_F / 512, ROWS / 64), dim3(256), 0, stream>>>(rbf, wbf, out);
}
